// RoPEAttention_3D_35321811042326
// MI455X (gfx1250) — matrix operand order repaired and checked by reading, not compiled, not run on hardware
//
#include <hip/hip_runtime.h>
#include <hip/hip_bf16.h>

// ---------------- problem constants ----------------
#define Bv   4
#define Nv   2048
#define Cv   1024
#define Hv   16
#define HDv  64
#define HALFv 32
#define TOK  (Bv * Nv)          // 8192 tokens
#define C3v  (3 * Cv)           // 3072

typedef __attribute__((ext_vector_type(16))) _Float16 v16h;
typedef __attribute__((ext_vector_type(8)))  float    v8f;

union FragU { v16h v; uint4 q[2]; };

// A-matrix fragment (16x32 f16, M x K): lane = (g = lane>>4, m = lane&15).
// Per lane: 8 halves at k = g*8 .. g*8+7 and 8 halves at k = 16+g*8 .. (ISA 7.12.2)
__device__ __forceinline__ v16h load_a16(const _Float16* p, int g) {
  FragU f;
  f.q[0] = *(const uint4*)(p + g * 8);
  f.q[1] = *(const uint4*)(p + 16 + g * 8);
  return f.v;
}

// B-matrix fragment (32x16 f16, K x N): lane = (g = lane>>4, n = lane&15).
// Per lane: 8 halves at k = g*8 .. g*8+7 and 8 halves at k = 16+g*8 .. (matches A).
__device__ __forceinline__ v16h load_b16(const _Float16* p, int g) {
  FragU f;
  f.q[0] = *(const uint4*)(p + g * 8);
  f.q[1] = *(const uint4*)(p + 16 + g * 8);
  return f.v;
}

__device__ __forceinline__ v8f wmma_f16(v16h a, v16h b, v8f c) {
  return __builtin_amdgcn_wmma_f32_16x16x32_f16(false, a, false, b, (short)0, c,
                                                false, false);
}

// Async copy 16 bytes/lane global -> LDS (GLOBAL_LOAD_ASYNC_TO_LDS_B128, ASYNCcnt).
// Generic __shared__ pointers carry the LDS offset in their low 32 bits.
__device__ __forceinline__ void async_cp16(void* lds_ptr, const void* gptr) {
  uint32_t loff = (uint32_t)(uintptr_t)lds_ptr;
  asm volatile("global_load_async_to_lds_b128 %0, %1, off"
               :: "v"(loff), "v"(gptr)
               : "memory");
}
__device__ __forceinline__ void wait_async0() {
  asm volatile("s_wait_asynccnt 0x0" ::: "memory");
}

// ---------------- kernel 1: min/max of t axes -> (min, VR/(max-min+eps)) ----------------
__global__ void kminmax(const float* __restrict__ tx, const float* __restrict__ ty,
                        const float* __restrict__ tz, float* __restrict__ nrm) {
  __shared__ float smin[256];
  __shared__ float smax[256];
  const float* arr[3] = {tx, ty, tz};
  for (int a = 0; a < 3; ++a) {
    float mn = 1e30f, mx = -1e30f;
    for (int i = threadIdx.x; i < Nv; i += 256) {
      float v = arr[a][i];
      mn = fminf(mn, v);
      mx = fmaxf(mx, v);
    }
    smin[threadIdx.x] = mn;
    smax[threadIdx.x] = mx;
    __syncthreads();
    for (int s = 128; s > 0; s >>= 1) {
      if (threadIdx.x < (unsigned)s) {
        smin[threadIdx.x] = fminf(smin[threadIdx.x], smin[threadIdx.x + s]);
        smax[threadIdx.x] = fmaxf(smax[threadIdx.x], smax[threadIdx.x + s]);
      }
      __syncthreads();
    }
    if (threadIdx.x == 0) {
      nrm[a * 2 + 0] = smin[0];
      nrm[a * 2 + 1] = 32.0f / (smax[0] - smin[0] + 1e-8f);
    }
    __syncthreads();
  }
}

// ---------------- kernel 2: cos/sin tables [H, N, HALF] ----------------
__global__ void kangles(const float* __restrict__ tx, const float* __restrict__ ty,
                        const float* __restrict__ tz, const float* __restrict__ freqs,
                        const float* __restrict__ nrm, float* __restrict__ cosT,
                        float* __restrict__ sinT) {
  int idx = blockIdx.x * blockDim.x + threadIdx.x;   // N * H * HALF = 1,048,576
  int r = idx & (Hv * HALFv - 1);                    // h*HALF + i
  int n = idx >> 9;
  float fx = (tx[n] - nrm[0]) * nrm[1];
  float fy = (ty[n] - nrm[2]) * nrm[3];
  float fz = (tz[n] - nrm[4]) * nrm[5];
  float a = fx * freqs[r] + fy * freqs[Hv * HALFv + r] + fz * freqs[2 * Hv * HALFv + r];
  int h = r >> 5;
  int i = r & (HALFv - 1);
  size_t o = ((size_t)h * Nv + n) * HALFv + i;
  cosT[o] = cosf(a);
  sinT[o] = sinf(a);
}

// ---------------- kernel 3: f32 -> f16 convert ----------------
__global__ void kcvt(const float* __restrict__ in, _Float16* __restrict__ out, int n) {
  int i = blockIdx.x * blockDim.x + threadIdx.x;
  if (i < n) out[i] = (_Float16)in[i];
}

// ---------------- kernel 4: QKV GEMM  (X[8192,1024] @ Wqkv[3072,1024]^T) ----------------
// Per wave: 32 rows x 64 cols (2 A-frags reuse each B-frag -> 8 WMMA / 12 b128 loads).
// Q,K -> fp32 [B,H,N,HD];  V -> f16 transposed [B,H,HD,N].
__global__ void kgemm_qkv(const _Float16* __restrict__ Xh, const _Float16* __restrict__ Wh,
                          float* __restrict__ Qf, float* __restrict__ Kf,
                          _Float16* __restrict__ Vt) {
  int w = blockIdx.x * 8 + (threadIdx.x >> 5);
  int lane = threadIdx.x & 31, g = lane >> 4, ln = lane & 15;
  int mt = w & 255;              // 256 row tiles of 32
  int nt = w >> 8;               // 48 col tiles of 64
  int mbase = mt * 32, nbase = nt * 64;
  const _Float16* arow0 = Xh + (size_t)(mbase + ln) * Cv;
  const _Float16* arow1 = Xh + (size_t)(mbase + 16 + ln) * Cv;
  v8f acc[2][4];
#pragma unroll
  for (int r = 0; r < 2; ++r)
#pragma unroll
    for (int c = 0; c < 4; ++c) acc[r][c] = v8f{};
  for (int kb = 0; kb < Cv; kb += 32) {
    v16h a0 = load_a16(arow0 + kb, g);
    v16h a1 = load_a16(arow1 + kb, g);
#pragma unroll
    for (int c = 0; c < 4; ++c) {
      v16h b = load_b16(Wh + (size_t)(nbase + c * 16 + ln) * Cv + kb, g);
      acc[0][c] = wmma_f16(a0, b, acc[0][c]);
      acc[1][c] = wmma_f16(a1, b, acc[1][c]);
    }
  }
#pragma unroll
  for (int r = 0; r < 2; ++r) {
#pragma unroll
    for (int c = 0; c < 4; ++c) {
#pragma unroll
      for (int v = 0; v < 8; ++v) {
        int m = mbase + r * 16 + g * 8 + v;   // token row
        int j = nbase + c * 16 + ln;          // output col in [0, 3072)
        int b = m >> 11, n = m & (Nv - 1);
        int which = j >> 10, jj = j & (Cv - 1);
        int h = jj >> 6, d = jj & (HDv - 1);
        float val = acc[r][c][v];
        if (which == 0) {
          Qf[(((size_t)(b * Hv + h) * Nv + n) * HDv) + d] = val;
        } else if (which == 1) {
          Kf[(((size_t)(b * Hv + h) * Nv + n) * HDv) + d] = val;
        } else {
          Vt[(((size_t)(b * Hv + h) * HDv + d) * Nv) + n] = (_Float16)val;
        }
      }
    }
  }
}

// ---------------- kernel 5: RoPE on Q,K pairs (fp32) -> f16 [B,H,N,HD] ----------------
__global__ void krope(const float* __restrict__ Qf, const float* __restrict__ Kf,
                      const float* __restrict__ cosT, const float* __restrict__ sinT,
                      _Float16* __restrict__ Qh, _Float16* __restrict__ Kh) {
  int idx = blockIdx.x * blockDim.x + threadIdx.x;   // B*H*N*HALF = 4,194,304
  int i = idx & (HALFv - 1);
  int n = (idx >> 5) & (Nv - 1);
  int h = (idx >> 16) & (Hv - 1);
  int b = idx >> 20;
  size_t base = (((size_t)(b * Hv + h) * Nv) + n) * HDv + 2 * i;
  size_t ti = ((size_t)h * Nv + n) * HALFv + i;
  float c = cosT[ti], s = sinT[ti];
  float a0 = Qf[base], b0 = Qf[base + 1];
  Qh[base]     = (_Float16)(a0 * c - b0 * s);
  Qh[base + 1] = (_Float16)(a0 * s + b0 * c);
  a0 = Kf[base];
  b0 = Kf[base + 1];
  Kh[base]     = (_Float16)(a0 * c - b0 * s);
  Kh[base + 1] = (_Float16)(a0 * s + b0 * c);
}

// ---------------- kernel 6: flash attention ----------------
// One wave per 16-query tile; all 8 waves of a block share (b,h) and the key loop,
// so K/V tiles are staged cooperatively into LDS with async-to-LDS copies.
__global__ void kattn(const _Float16* __restrict__ Qh, const _Float16* __restrict__ Kh,
                      const _Float16* __restrict__ Vt, _Float16* __restrict__ AOb) {
  __shared__ __align__(16) _Float16 Ktile[32 * 64];    // [key][d]   4 KB
  __shared__ __align__(16) _Float16 Vtile[64 * 32];    // [d][key]   4 KB
  __shared__ __align__(16) _Float16 Pld[8][16 * 32];   // per-wave P 8 KB
  int tid = threadIdx.x;
  int wv = tid >> 5;
  int lane = tid & 31, g = lane >> 4, ln = lane & 15;
  int w = blockIdx.x * 8 + wv;
  int qt = w & (Nv / 16 - 1);     // 128 query tiles (8 consecutive per block)
  int bh = w >> 7;                // b*H + h (uniform across block)
  int qbase = qt * 16;

  const _Float16* qrow = Qh + ((size_t)bh * Nv + qbase + ln) * HDv;
  v16h aq0 = load_a16(qrow, g);         // head dims 0..31
  v16h aq1 = load_a16(qrow + 32, g);    // head dims 32..63

  v8f o[4] = {v8f{}, v8f{}, v8f{}, v8f{}};
  float mrow[8], lrow[8];
#pragma unroll
  for (int v = 0; v < 8; ++v) { mrow[v] = -1e30f; lrow[v] = 0.0f; }

  const _Float16* kbasep = Kh + (size_t)bh * Nv * HDv;
  const _Float16* vbasep = Vt + (size_t)bh * HDv * Nv;
  _Float16* P = &Pld[wv][0];
  const float scale = 0.125f;   // HD^-0.5

  for (int kb = 0; kb < Nv; kb += 32) {
    // --- cooperative async staging: 256 threads x 16B per tile ---
    async_cp16(Ktile + tid * 8,
               kbasep + (size_t)(kb + (tid >> 3)) * HDv + (tid & 7) * 8);
    async_cp16(Vtile + tid * 8,
               vbasep + (size_t)(tid >> 2) * Nv + kb + (tid & 3) * 8);
    wait_async0();
    __syncthreads();

    v8f s[2];
#pragma unroll
    for (int sub = 0; sub < 2; ++sub) {
      const _Float16* krow = Ktile + (size_t)(sub * 16 + ln) * HDv;
      v16h b0 = load_b16(krow, g);
      v16h b1 = load_b16(krow + 32, g);
      v8f z = {};
      z = wmma_f16(aq0, b0, z);
      z = wmma_f16(aq1, b1, z);
      s[sub] = z;
    }
    float alpha[8];
#pragma unroll
    for (int v = 0; v < 8; ++v) {
      float s0 = s[0][v] * scale;
      float s1 = s[1][v] * scale;
      float tm = fmaxf(s0, s1);
      tm = fmaxf(tm, __shfl_xor(tm, 1, 32));
      tm = fmaxf(tm, __shfl_xor(tm, 2, 32));
      tm = fmaxf(tm, __shfl_xor(tm, 4, 32));
      tm = fmaxf(tm, __shfl_xor(tm, 8, 32));
      float nm = fmaxf(mrow[v], tm);
      float al = __expf(mrow[v] - nm);
      float p0 = __expf(s0 - nm);
      float p1 = __expf(s1 - nm);
      float rs = p0 + p1;
      rs += __shfl_xor(rs, 1, 32);
      rs += __shfl_xor(rs, 2, 32);
      rs += __shfl_xor(rs, 4, 32);
      rs += __shfl_xor(rs, 8, 32);
      lrow[v] = lrow[v] * al + rs;
      mrow[v] = nm;
      alpha[v] = al;
      int row = g * 8 + v;
      P[row * 32 + ln]      = (_Float16)p0;
      P[row * 32 + 16 + ln] = (_Float16)p1;
    }
    __syncthreads();
#pragma unroll
    for (int dt = 0; dt < 4; ++dt)
#pragma unroll
      for (int v = 0; v < 8; ++v) o[dt][v] *= alpha[v];
    v16h ap = load_a16(P + ln * 32, g);   // A frag of P: 16 rows x 32 keys (LDS)
#pragma unroll
    for (int dt = 0; dt < 4; ++dt) {
      v16h bv = load_b16(Vtile + (size_t)(dt * 16 + ln) * 32, g);
      o[dt] = wmma_f16(ap, bv, o[dt]);
    }
    __syncthreads();   // protect Ktile/Vtile/P before next staging
  }

  int b = bh >> 4, h = bh & (Hv - 1);
#pragma unroll
  for (int v = 0; v < 8; ++v) {
    float inv = 1.0f / lrow[v];
    size_t tok = (size_t)b * Nv + qbase + g * 8 + v;
    size_t off = tok * Cv + h * HDv;
#pragma unroll
    for (int dt = 0; dt < 4; ++dt)
      AOb[off + dt * 16 + ln] = (_Float16)(o[dt][v] * inv);
  }
}

// ---------------- kernel 7: output projection + bias ----------------
__global__ void kgemm_proj(const _Float16* __restrict__ AO, const _Float16* __restrict__ Wh,
                           const float* __restrict__ bias, float* __restrict__ out) {
  int w = blockIdx.x * 8 + (threadIdx.x >> 5);
  int lane = threadIdx.x & 31, g = lane >> 4, ln = lane & 15;
  int mt = w & 255;               // 256 row tiles of 32
  int nt = w >> 8;                // 16 col tiles of 64
  int mbase = mt * 32, nbase = nt * 64;
  const _Float16* arow0 = AO + (size_t)(mbase + ln) * Cv;
  const _Float16* arow1 = AO + (size_t)(mbase + 16 + ln) * Cv;
  v8f acc[2][4];
#pragma unroll
  for (int r = 0; r < 2; ++r)
#pragma unroll
    for (int c = 0; c < 4; ++c) acc[r][c] = v8f{};
  for (int kb = 0; kb < Cv; kb += 32) {
    v16h a0 = load_a16(arow0 + kb, g);
    v16h a1 = load_a16(arow1 + kb, g);
#pragma unroll
    for (int c = 0; c < 4; ++c) {
      v16h b = load_b16(Wh + (size_t)(nbase + c * 16 + ln) * Cv + kb, g);
      acc[0][c] = wmma_f16(a0, b, acc[0][c]);
      acc[1][c] = wmma_f16(a1, b, acc[1][c]);
    }
  }
#pragma unroll
  for (int r = 0; r < 2; ++r) {
#pragma unroll
    for (int c = 0; c < 4; ++c) {
#pragma unroll
      for (int v = 0; v < 8; ++v) {
        int m = mbase + r * 16 + g * 8 + v;
        int j = nbase + c * 16 + ln;
        out[(size_t)m * Cv + j] = acc[r][c][v] + bias[j];
      }
    }
  }
}

// ---------------- host launch ----------------
extern "C" void kernel_launch(void* const* d_in, const int* in_sizes, int n_in,
                              void* d_out, int out_size, void* d_ws, size_t ws_size,
                              hipStream_t stream) {
  (void)in_sizes; (void)n_in; (void)out_size; (void)ws_size;
  const float* x      = (const float*)d_in[0];
  const float* t_x    = (const float*)d_in[1];
  const float* t_y    = (const float*)d_in[2];
  const float* t_z    = (const float*)d_in[3];
  const float* w_qkv  = (const float*)d_in[4];
  const float* w_proj = (const float*)d_in[5];
  const float* b_proj = (const float*)d_in[6];
  const float* freqs  = (const float*)d_in[7];
  float* out = (float*)d_out;

  // workspace carve-up (all offsets 256B aligned)
  char* p = (char*)d_ws;
  auto take = [&](size_t bytes) -> char* {
    char* r = p;
    p += (bytes + 255) & ~(size_t)255;
    return r;
  };
  float*     nrm    = (float*)take(64);
  float*     cosT   = (float*)take((size_t)Hv * Nv * HALFv * 4);          // 4 MB
  float*     sinT   = (float*)take((size_t)Hv * Nv * HALFv * 4);          // 4 MB
  float*     Qf     = (float*)take((size_t)TOK * Cv * 4);                 // 32 MB
  float*     Kf     = (float*)take((size_t)TOK * Cv * 4);                 // 32 MB
  _Float16*  Xh     = (_Float16*)take((size_t)TOK * Cv * 2);              // 16 MB
  _Float16*  Wqkvh  = (_Float16*)take((size_t)C3v * Cv * 2);              // 6 MB
  _Float16*  Wprojh = (_Float16*)take((size_t)Cv * Cv * 2);               // 2 MB
  _Float16*  Qh     = (_Float16*)take((size_t)TOK * Cv * 2);              // 16 MB
  _Float16*  Kh     = (_Float16*)take((size_t)TOK * Cv * 2);              // 16 MB
  _Float16*  Vt     = (_Float16*)take((size_t)TOK * Cv * 2);              // 16 MB
  _Float16*  AOb    = (_Float16*)take((size_t)TOK * Cv * 2);              // 16 MB

  kminmax<<<1, 256, 0, stream>>>(t_x, t_y, t_z, nrm);
  kangles<<<(Nv * Hv * HALFv) / 256, 256, 0, stream>>>(t_x, t_y, t_z, freqs, nrm,
                                                       cosT, sinT);
  kcvt<<<(TOK * Cv) / 256, 256, 0, stream>>>(x, Xh, TOK * Cv);
  kcvt<<<(C3v * Cv) / 256, 256, 0, stream>>>(w_qkv, Wqkvh, C3v * Cv);
  kcvt<<<(Cv * Cv) / 256, 256, 0, stream>>>(w_proj, Wprojh, Cv * Cv);

  // 256 row-tiles x 48 col-tiles = 12288 waves / 8 per block
  kgemm_qkv<<<1536, 256, 0, stream>>>(Xh, Wqkvh, Qf, Kf, Vt);

  krope<<<(Bv * Hv * Nv * HALFv) / 256, 256, 0, stream>>>(Qf, Kf, cosT, sinT, Qh, Kh);

  // B*H*(N/16) = 8192 waves / 8 per block
  kattn<<<1024, 256, 0, stream>>>(Qh, Kh, Vt, AOb);

  // 256 x 16 = 4096 waves / 8 per block
  kgemm_proj<<<512, 256, 0, stream>>>(AOb, Wprojh, b_proj, out);
}
